// TensorProductConvLayer_18726057411380
// MI455X (gfx1250) — matrix operand order repaired and checked by reading, not compiled, not run on hardware
//
#include <hip/hip_runtime.h>
#include <math.h>

#define N_NODES   50000
#define N_EDGES   400000
#define MUL0      16
#define MUL1      8
#define IN_DIM    40        // MUL0 + 3*MUL1
#define EDGE_FEAT 32
#define WNUM      576       // 256 + 64 + 128 + 128

typedef _Float16 v16h __attribute__((ext_vector_type(16)));
typedef _Float16 v8h  __attribute__((ext_vector_type(8)));
typedef float    v8f  __attribute__((ext_vector_type(8)));

__device__ __forceinline__ v8f wmma_f16(v16h a, v16h b, v8f c) {
  // D = A(16x32,f16) x B(32x16,f16) + C(16x16,f32)
  return __builtin_amdgcn_wmma_f32_16x16x32_f16(false, a, false, b, (short)0, c,
                                                false, false);
}

// Branch-free fast softplus: max(x,0) + log(1 + exp(-|x|)).
// exp argument is <= 0 so no overflow; uses v_exp_f32/v_log_f32 hw trans ops.
__device__ __forceinline__ float fast_softplus(float x) {
  float e = __expf(-fabsf(x));
  return fmaxf(x, 0.0f) + __logf(1.0f + e);
}

__device__ __constant__ float kInvSqrt3 = 0.57735026918962576f;
__device__ __constant__ float kAlpha0   = 0.20412414523193148f; // sqrt(1/24)
__device__ __constant__ float kAlpha1   = 0.35355339059327373f; // sqrt(3/24)

// ---- LDS layout (dynamic) ------------------------------------------------
// [0, 36864)           : w2 B-fragments, 36 tiles x 32 lanes x 16 halves (f16)
// [36864, 36864+4096)  : per-wave 16x32 f16 h-tiles (4 waves x 1024 B)
// [40960, 49664)       : per-wave double-buffered 16x17 f32 D-tile (4 x 2176 B)
#define LDS_W2_HALVES   (36 * 512)
#define LDS_W2_BYTES    (LDS_W2_HALVES * 2)
#define LDS_H_PER_WAVE  (16 * 32)          // halves
#define LDS_T_PER_BUF   (16 * 17)          // floats, padded stride 17
#define WAVES_PER_BLOCK 4
#define BLOCK_THREADS   (WAVES_PER_BLOCK * 32)
#define EDGES_PER_BLOCK (WAVES_PER_BLOCK * 16)
#define SMEM_BYTES (LDS_W2_BYTES + WAVES_PER_BLOCK * LDS_H_PER_WAVE * 2 + \
                    WAVES_PER_BLOCK * 2 * LDS_T_PER_BUF * 4)

__global__ __launch_bounds__(BLOCK_THREADS)
void tp_conv_kernel(const float* __restrict__ node_attr,
                    const int*   __restrict__ edge_index,
                    const float* __restrict__ edge_attr,
                    const float* __restrict__ edge_sh,
                    const float* __restrict__ w1g,
                    const float* __restrict__ b1g,
                    const float* __restrict__ w2g,
                    const float* __restrict__ b2g,
                    float* __restrict__ outSum,
                    float* __restrict__ counts)
{
  extern __shared__ char smem[];
  _Float16* sW2 = (_Float16*)smem;

  const int tid  = threadIdx.x;
  const int wave = tid >> 5;
  const int lane = tid & 31;
  const int eLoc = lane & 15;   // edge row within the 16-edge tile
  const int half = lane >> 4;   // 0: lanes 0-15, 1: lanes 16-31

  _Float16* sH = (_Float16*)(smem + LDS_W2_BYTES) + wave * LDS_H_PER_WAVE;
  float*    sT = (float*)(smem + LDS_W2_BYTES + WAVES_PER_BLOCK * LDS_H_PER_WAVE * 2)
                 + wave * (2 * LDS_T_PER_BUF);

  // --- stage w2 (32x576 f32) into LDS as f16 B-fragments, fragment order ---
  // idx = t*512 + lane*16 + h  holds  w2[k, col], col=t*16+(lane&15),
// k=(lane>=16 ? 8:0)+(h&7)+(h>=8 ? 16:0)   (matches ISA 16-bit B layout)
  for (int idx = tid; idx < LDS_W2_HALVES; idx += BLOCK_THREADS) {
    int t = idx >> 9;
    int r = idx & 511;
    int l = r >> 4;
    int h = r & 15;
    int col = (t << 4) + (l & 15);
    int k   = ((l >> 4) << 3) + (h & 7) + ((h & 8) << 1);
    sW2[idx] = (_Float16)w2g[k * WNUM + col];
  }
  __syncthreads();

  const int eG    = blockIdx.x * EDGES_PER_BLOCK + wave * 16 + eLoc;
  const bool valid = (eG < N_EDGES);
  const int eC    = valid ? eG : (N_EDGES - 1);

  // --- GEMM1 A fragment: edge_attr tile (16 edges x 32 feats) -> f16 -------
  // lane layout: row=eLoc; halves 0..7 -> K=kb..kb+7, 8..15 -> K=kb+16..kb+23
  const float* ea = edge_attr + eC * EDGE_FEAT;
  const int kbA = half ? 8 : 0;
  v16h a1;
  {
    float4 f0 = *(const float4*)(ea + kbA);
    float4 f1 = *(const float4*)(ea + kbA + 4);
    float4 f2 = *(const float4*)(ea + kbA + 16);
    float4 f3 = *(const float4*)(ea + kbA + 20);
    a1[0]=(_Float16)f0.x;  a1[1]=(_Float16)f0.y;  a1[2]=(_Float16)f0.z;  a1[3]=(_Float16)f0.w;
    a1[4]=(_Float16)f1.x;  a1[5]=(_Float16)f1.y;  a1[6]=(_Float16)f1.z;  a1[7]=(_Float16)f1.w;
    a1[8]=(_Float16)f2.x;  a1[9]=(_Float16)f2.y;  a1[10]=(_Float16)f2.z; a1[11]=(_Float16)f2.w;
    a1[12]=(_Float16)f3.x; a1[13]=(_Float16)f3.y; a1[14]=(_Float16)f3.z; a1[15]=(_Float16)f3.w;
  }

  // --- GEMM1 B fragments from w1 (32x32), two N-tiles ----------------------
  const int kbB  = half << 3;
  const int colA = eLoc;
  const int colB = 16 + eLoc;
  v16h bA, bB;
  #pragma unroll
  for (int h = 0; h < 16; ++h) {
    bA[h] = (_Float16)w1g[(kbB + (h & 7) + ((h & 8) << 1)) * 32 + colA];
    bB[h] = (_Float16)w1g[(kbB + (h & 7) + ((h & 8) << 1)) * 32 + colB];
  }
  v8f cA, cB;
  {
    float biasA = b1g[colA], biasB = b1g[colB];
    #pragma unroll
    for (int v = 0; v < 8; ++v) { cA[v] = biasA; cB[v] = biasB; }
  }

  v8f hA = wmma_f16(a1, bA, cA);
  v8f hB = wmma_f16(a1, bB, cB);

  // softplus + f16, write h-tile to LDS (row-major 16x32 halves)
  #pragma unroll
  for (int v = 0; v < 8; ++v) {
    int row = v + 8 * half;
    float sa = fast_softplus(hA[v]);
    float sb = fast_softplus(hB[v]);
    sH[row * 32 + colA] = (_Float16)sa;
    sH[row * 32 + colB] = (_Float16)sb;
  }

  // --- GEMM2 A fragment (h tile, 16x32) via two 16B LDS vector loads -------
  v16h a2;
  {
    const int kb2 = half ? 8 : 0;
    v8h lo  = *(const v8h*)(sH + eLoc * 32 + kb2);
    v8h hi8 = *(const v8h*)(sH + eLoc * 32 + kb2 + 16);
    #pragma unroll
    for (int i = 0; i < 8; ++i) { a2[i] = lo[i]; a2[8 + i] = hi8[i]; }
  }

  // --- per-edge data for the tensor product --------------------------------
  const int src = edge_index[eC];
  const int dst = edge_index[N_EDGES + eC];
  const float y0  = edge_sh[eC * 4 + 0];
  const float y1x = edge_sh[eC * 4 + 1];
  const float y1y = edge_sh[eC * 4 + 2];
  const float y1z = edge_sh[eC * 4 + 3];
  const float* x = node_attr + dst * IN_DIM;   // s0 = x[0..15], v1 = x[16..39]

  float acc0[16];
  float g[8], px[8], py[8], pz[8];
  #pragma unroll
  for (int i = 0; i < 16; ++i) acc0[i] = 0.0f;
  #pragma unroll
  for (int i = 0; i < 8; ++i) { g[i] = 0.0f; px[i] = 0.0f; py[i] = 0.0f; pz[i] = 0.0f; }

  // one 16-col tile of w = h @ w2 + b2, written to padded LDS tile buffer
  auto computeTile = [&](int t, float* tb) {
    const _Float16* bp = sW2 + t * 512 + lane * 16;
    v8h blo = *(const v8h*)bp;
    v8h bhi = *(const v8h*)(bp + 8);
    v16h bf;
    #pragma unroll
    for (int i = 0; i < 8; ++i) { bf[i] = blo[i]; bf[8 + i] = bhi[i]; }
    float bias = b2g[(t << 4) + eLoc];
    v8f c;
    #pragma unroll
    for (int v = 0; v < 8; ++v) c[v] = bias;
    v8f d = wmma_f16(a2, bf, c);
    #pragma unroll
    for (int v = 0; v < 8; ++v) tb[(v + 8 * half) * 17 + eLoc] = d[v];
  };

  // w00 region: tiles 0..15, u = t      (out0 += s0[u]*y0 * w00[u,:])
  for (int t = 0; t < 16; ++t) {
    float* tb = sT + (t & 1) * LDS_T_PER_BUF;
    computeTile(t, tb);
    if (!half) {
      float a0u = x[t] * y0;
      const float* row = tb + eLoc * 17;
      #pragma unroll
      for (int w = 0; w < 16; ++w) acc0[w] += a0u * row[w];
    }
  }
  // w10 region: tiles 16..19 (u pairs)  (p[w,m] += v1[u,m] * w10[u,w])
  for (int t = 16; t < 20; ++t) {
    float* tb = sT + (t & 1) * LDS_T_PER_BUF;
    computeTile(t, tb);
    if (half) {
      int u0 = (t - 16) * 2;
      const float* row = tb + eLoc * 17;
      #pragma unroll
      for (int s = 0; s < 2; ++s) {
        int u = u0 + s;
        float vx = x[16 + 3 * u], vy = x[17 + 3 * u], vz = x[18 + 3 * u];
        #pragma unroll
        for (int w = 0; w < 8; ++w) {
          float wv = row[s * 8 + w];
          px[w] += vx * wv; py[w] += vy * wv; pz[w] += vz * wv;
        }
      }
    }
  }
  // w01 region: tiles 20..27 (u pairs)  (g[w] += s0[u] * w01[u,w])
  for (int t = 20; t < 28; ++t) {
    float* tb = sT + (t & 1) * LDS_T_PER_BUF;
    computeTile(t, tb);
    if (half) {
      int u0 = (t - 20) * 2;
      const float* row = tb + eLoc * 17;
      #pragma unroll
      for (int s = 0; s < 2; ++s) {
        float s0u = x[u0 + s];
        #pragma unroll
        for (int w = 0; w < 8; ++w) g[w] += s0u * row[s * 8 + w];
      }
    }
  }
  // w11 region: tiles 28..35, u = t-28  (out0 += (v1[u]·y1)/sqrt3 * w11[u,:])
  for (int t = 28; t < 36; ++t) {
    float* tb = sT + (t & 1) * LDS_T_PER_BUF;
    computeTile(t, tb);
    if (!half) {
      int u = t - 28;
      float c1u = kInvSqrt3 *
          (x[16 + 3 * u] * y1x + x[17 + 3 * u] * y1y + x[18 + 3 * u] * y1z);
      const float* row = tb + eLoc * 17;
      #pragma unroll
      for (int w = 0; w < 16; ++w) acc0[w] += c1u * row[w];
    }
  }

  // --- scatter-add (segment sum) -------------------------------------------
  if (valid) {
    float* o = outSum + src * IN_DIM;
    if (!half) {
      #pragma unroll
      for (int w = 0; w < 16; ++w) atomicAdd(&o[w], kAlpha0 * acc0[w]);
      atomicAdd(&counts[src], 1.0f);
    } else {
      const float A1 = kAlpha1 * kInvSqrt3;
      #pragma unroll
      for (int w = 0; w < 8; ++w) {
        atomicAdd(&o[16 + 3 * w + 0], A1 * (y0 * px[w] + y1x * g[w]));
        atomicAdd(&o[16 + 3 * w + 1], A1 * (y0 * py[w] + y1y * g[w]));
        atomicAdd(&o[16 + 3 * w + 2], A1 * (y0 * pz[w] + y1z * g[w]));
      }
    }
  }
}

__global__ __launch_bounds__(256)
void finalize_kernel(const float* __restrict__ node_attr,
                     const float* __restrict__ counts,
                     float* __restrict__ out)
{
  int i = blockIdx.x * blockDim.x + threadIdx.x;
  if (i < N_NODES * IN_DIM) {
    int n = i / IN_DIM;
    float c = counts[n];
    c = (c < 1.0f) ? 1.0f : c;
    out[i] = out[i] / c + node_attr[i];
  }
}

extern "C" void kernel_launch(void* const* d_in, const int* in_sizes, int n_in,
                              void* d_out, int out_size, void* d_ws, size_t ws_size,
                              hipStream_t stream) {
  const float* node_attr = (const float*)d_in[0];
  const int*   edge_index = (const int*)d_in[1];
  const float* edge_attr = (const float*)d_in[2];
  const float* edge_sh   = (const float*)d_in[3];
  const float* w1 = (const float*)d_in[4];
  const float* b1 = (const float*)d_in[5];
  const float* w2 = (const float*)d_in[6];
  const float* b2 = (const float*)d_in[7];

  float* out    = (float*)d_out;
  float* counts = (float*)d_ws;           // N_NODES floats of scratch

  hipMemsetAsync(d_out, 0, (size_t)N_NODES * IN_DIM * sizeof(float), stream);
  hipMemsetAsync(d_ws, 0, (size_t)N_NODES * sizeof(float), stream);

  dim3 grid1((N_EDGES + EDGES_PER_BLOCK - 1) / EDGES_PER_BLOCK);
  tp_conv_kernel<<<grid1, BLOCK_THREADS, SMEM_BYTES, stream>>>(
      node_attr, edge_index, edge_attr, edge_sh, w1, b1, w2, b2, out, counts);

  dim3 grid2((N_NODES * IN_DIM + 255) / 256);
  finalize_kernel<<<grid2, 256, 0, stream>>>(node_attr, counts, out);
}
